// EncoderLayer_26534307954754
// MI455X (gfx1250) — matrix operand order repaired and checked by reading, not compiled, not run on hardware
//
#include <hip/hip_runtime.h>
#include <math.h>
#include <stdint.h>

// ---------------------------------------------------------------------------
// Problem constants (match the reference)
// ---------------------------------------------------------------------------
#define BB   16
#define PP   1024
#define DM   512
#define HH   8
#define ADIM 64
#define FFD  2048
#define MM   (BB * PP)          // 16384 rows
#define NX   (MM * DM)          // 8388608 elements in x
#define QK_SCALE 0.125f         // 1/sqrt(64)
#define BSTRIDE 40              // padded LDS row stride (elems): 80B, 16B-aligned,
                                // 20-dword stride -> 16 distinct banks across lanes

typedef __attribute__((ext_vector_type(16))) __bf16 v16bf;
typedef __attribute__((ext_vector_type(8)))  __bf16 v8bf;
typedef __attribute__((ext_vector_type(8)))  float  v8f;

// gfx1250 async global->LDS path (ASYNCcnt-tracked), with portable fallback.
// Builtin signature (per hipcc diagnostic): (int4 AS1*, int4 AS3*, imm, imm).
#if defined(__HIP_DEVICE_COMPILE__) && defined(__gfx1250__) &&        \
    __has_builtin(__builtin_amdgcn_global_load_async_to_lds_b128) &&  \
    __has_builtin(__builtin_amdgcn_s_wait_asynccnt)
#define USE_ASYNC_LDS 1
typedef __attribute__((vector_size(4 * sizeof(int)))) int i4t;
typedef __attribute__((address_space(1))) i4t GI4;
typedef __attribute__((address_space(3))) i4t LI4;
#else
#define USE_ASYNC_LDS 0
#endif

__device__ __forceinline__ void wait_async_le2() {
#if USE_ASYNC_LDS
  __builtin_amdgcn_s_wait_asynccnt(2);   // retire the older stage (in-order)
#endif
}
__device__ __forceinline__ void wait_async_le0() {
#if USE_ASYNC_LDS
  __builtin_amdgcn_s_wait_asynccnt(0);
#endif
}

// ---------------------------------------------------------------------------
// WMMA helpers (CDNA5 wave32 layouts, bf16 A/B, f32 accumulate)
// ---------------------------------------------------------------------------
__device__ __forceinline__ v8f wmma_bf16(v16bf a, v16bf b, v8f c) {
  return __builtin_amdgcn_wmma_f32_16x16x32_bf16(
      /*neg_a=*/false, a, /*neg_b=*/false, b,
      /*c_mod=*/(short)0, c, /*reuse_a=*/false, /*reuse_b=*/false);
}

// A-fragment: 16(M) x 32(K) bf16, row-major source, leading dim `ld`.
// Lane L: row = L%16, kbase = (L/16)*8; elems 0..7 -> K=kbase..kbase+7,
// elems 8..15 -> K=kbase+16..kbase+23  (ISA 16-bit A layout).
__device__ __forceinline__ v16bf load_a_frag(const __bf16* base, int ld) {
  int lane = threadIdx.x & 31;
  int row  = lane & 15;
  int kb   = (lane >> 4) * 8;
  const __bf16* p = base + (size_t)row * ld + kb;
  v8bf lo = *(const v8bf*)(p);
  v8bf hi = *(const v8bf*)(p + 16);
  v16bf a;
#pragma unroll
  for (int i = 0; i < 8; ++i) { a[i] = lo[i]; a[i + 8] = hi[i]; }
  return a;
}

// B-fragment from (N,K)-major global memory, leading dim `ld`.
// Lane L: col = L%16, kbase = (L/16)*8; elems 0..7 -> K=kbase..kbase+7,
// elems 8..15 -> K=kbase+16..kbase+23 (matches A layout).
__device__ __forceinline__ v16bf load_b_frag(const __bf16* baseNK, int ld) {
  int lane = threadIdx.x & 31;
  int col  = lane & 15;
  int kb   = (lane >> 4) * 8;
  const __bf16* p = baseNK + (size_t)col * ld + kb;
  v8bf lo = *(const v8bf*)(p);
  v8bf hi = *(const v8bf*)(p + 16);
  v16bf b;
#pragma unroll
  for (int i = 0; i < 8; ++i) { b[i] = lo[i]; b[i + 8] = hi[i]; }
  return b;
}

// B-fragment from an LDS-staged tile (row stride BSTRIDE elems).
__device__ __forceinline__ v16bf load_b_frag_lds(const __bf16* base) {
  int lane = threadIdx.x & 31;
  int col  = lane & 15;
  int kb   = (lane >> 4) * 8;
  const __bf16* p = base + (size_t)col * BSTRIDE + kb;
  v8bf lo = *(const v8bf*)(p);
  v8bf hi = *(const v8bf*)(p + 16);
  v16bf b;
#pragma unroll
  for (int i = 0; i < 8; ++i) { b[i] = lo[i]; b[i + 8] = hi[i]; }
  return b;
}

__device__ __forceinline__ float gelu_exact(float v) {
  return 0.5f * v * (1.0f + erff(v * 0.70710678118654752f));
}

// ---------------------------------------------------------------------------
// Async-staged, double-buffered shared-B GEMM core.
// Block = 128 threads (4 waves); tile = 64(M) x 64(N); per K-step the 64x32
// B tile is DMA'd to LDS once and consumed by all 4 waves.
// ---------------------------------------------------------------------------
__device__ __forceinline__ void stage_b_tile(const __bf16* Bt, size_t ldb,
                                             size_t n0, int kk,
                                             __bf16 (*buf)[BSTRIDE]) {
  int t = threadIdx.x;        // 0..127
  int n = t >> 1;             // 0..63 : tile row (N index)
  int half = t & 1;           // which 16-elem half of the 32-elem row
  const __bf16* src = Bt + (n0 + n) * ldb + (size_t)kk * 32 + half * 16;
  __bf16* dst = &buf[n][half * 16];
#if USE_ASYNC_LDS
  __builtin_amdgcn_global_load_async_to_lds_b128((GI4*)src, (LI4*)dst, 0, 0);
  __builtin_amdgcn_global_load_async_to_lds_b128((GI4*)(src + 8),
                                                 (LI4*)(dst + 8), 0, 0);
#else
  *(v8bf*)dst = *(const v8bf*)src;
  *(v8bf*)(dst + 8) = *(const v8bf*)(src + 8);
#endif
}

__device__ __forceinline__ void staged_gemm_16x64(
    const __bf16* __restrict__ A, size_t lda, const __bf16* __restrict__ Bt,
    size_t ldb, size_t n0, size_t m0, int KC, __bf16 (*Bs)[64][BSTRIDE],
    v8f acc[4]) {
  stage_b_tile(Bt, ldb, n0, 0, Bs[0]);
  for (int kk = 0; kk < KC; ++kk) {        // uniform control flow (EXEC full)
    v16bf a = load_a_frag(A + m0 * lda + (size_t)kk * 32, (int)lda);
    if (kk + 1 < KC) {
      stage_b_tile(Bt, ldb, n0, kk + 1, Bs[(kk + 1) & 1]);  // prefetch next
      wait_async_le2();                     // stage kk complete (in-order)
    } else {
      wait_async_le0();
    }
    __syncthreads();                        // stage kk visible to all waves
    const __bf16* bb = &Bs[kk & 1][0][0];
#pragma unroll
    for (int n = 0; n < 4; ++n)
      acc[n] = wmma_bf16(a, load_b_frag_lds(bb + (size_t)(n * 16) * BSTRIDE),
                         acc[n]);
    __syncthreads();                        // done reading buf before reuse
  }
}

// ---------------------------------------------------------------------------
// Stage 0: deterministic two-stage global sum / sum-of-squares
// ---------------------------------------------------------------------------
__global__ __launch_bounds__(256) void reduce_partial_kernel(
    const float* __restrict__ x, int n, float* __restrict__ psum,
    float* __restrict__ psq) {
  __shared__ float ls[256], lq[256];
  float s = 0.f, q = 0.f;
  for (int i = blockIdx.x * 256 + threadIdx.x; i < n; i += gridDim.x * 256) {
    float v = x[i];
    s += v;
    q += v * v;
  }
  ls[threadIdx.x] = s;
  lq[threadIdx.x] = q;
  __syncthreads();
  for (int off = 128; off > 0; off >>= 1) {
    if ((int)threadIdx.x < off) {
      ls[threadIdx.x] += ls[threadIdx.x + off];
      lq[threadIdx.x] += lq[threadIdx.x + off];
    }
    __syncthreads();
  }
  if (threadIdx.x == 0) {
    psum[blockIdx.x] = ls[0];
    psq[blockIdx.x]  = lq[0];
  }
}

__global__ __launch_bounds__(256) void finalize_stats_kernel(
    const float* __restrict__ psum, const float* __restrict__ psq, int nb,
    float n, float* __restrict__ stats) {
  __shared__ float ls[256], lq[256];
  float s = 0.f, q = 0.f;
  for (int i = threadIdx.x; i < nb; i += 256) { s += psum[i]; q += psq[i]; }
  ls[threadIdx.x] = s;
  lq[threadIdx.x] = q;
  __syncthreads();
  for (int off = 128; off > 0; off >>= 1) {
    if ((int)threadIdx.x < off) {
      ls[threadIdx.x] += ls[threadIdx.x + off];
      lq[threadIdx.x] += lq[threadIdx.x + off];
    }
    __syncthreads();
  }
  if (threadIdx.x == 0) {
    float mean = ls[0] / n;
    float var  = (lq[0] - n * mean * mean) / (n - 1.0f);  // unbiased
    stats[0] = mean;
    stats[1] = rsqrtf(var);
  }
}

__global__ __launch_bounds__(256) void norm_to_bf16_kernel(
    const float* __restrict__ x, const float* __restrict__ stats,
    const float* __restrict__ gamma, const float* __restrict__ beta,
    __bf16* __restrict__ out, int n) {
  float mean = stats[0], rstd = stats[1], g = gamma[0], b = beta[0];
  for (int i = blockIdx.x * 256 + threadIdx.x; i < n; i += gridDim.x * 256)
    out[i] = (__bf16)(g * (x[i] - mean) * rstd + b);
}

// Transpose-convert f32 (batch,R,C) -> bf16 (batch,C,R)
__global__ __launch_bounds__(256) void transpose_to_bf16_kernel(
    const float* __restrict__ in, __bf16* __restrict__ out, int batch, int R,
    int C) {
  int total = batch * R * C;
  for (int i = blockIdx.x * 256 + threadIdx.x; i < total;
       i += gridDim.x * 256) {
    int b = i / (R * C);
    int r = (i / C) % R;
    int c = i % C;
    out[((size_t)b * C + c) * R + r] = (__bf16)in[i];
  }
}

// ---------------------------------------------------------------------------
// Stage 1: QKV projection.  xn(16384x512) @ W(512x64) per head.
// grid = (MM/64, H, 3 {q,k,v}), block = 128.  B tile async-staged in LDS.
// Q,K stored (h,b,p,ad) [scale folded into Q]; V stored transposed (h,b,ad,p).
// ---------------------------------------------------------------------------
__global__ __launch_bounds__(128) void qkv_proj_kernel(
    const __bf16* __restrict__ xn, const __bf16* __restrict__ wqt,
    const __bf16* __restrict__ wkt, const __bf16* __restrict__ wvt,
    __bf16* __restrict__ Qo, __bf16* __restrict__ Ko,
    __bf16* __restrict__ Vt) {
  __shared__ __bf16 Bs[2][64][BSTRIDE];
  int wave = threadIdx.x >> 5, lane = threadIdx.x & 31;
  size_t m0 = (size_t)blockIdx.x * 64 + wave * 16;
  int h = blockIdx.y;
  int which = blockIdx.z;  // uniform per block
  const __bf16* wt =
      (which == 0 ? wqt : which == 1 ? wkt : wvt) + (size_t)h * ADIM * DM;

  v8f acc[4] = {};
  staged_gemm_16x64(xn, DM, wt, DM, /*n0=*/0, m0, /*KC=*/16, Bs, acc);

  int b = (int)(m0 >> 10), p = (int)(m0 & (PP - 1));
  int col = lane & 15, rb = (lane >> 4) * 8;
  if (which < 2) {
    float scale = (which == 0) ? QK_SCALE : 1.0f;
    __bf16* out =
        (which == 0 ? Qo : Ko) + ((size_t)(h * BB + b) * PP + p) * ADIM;
#pragma unroll
    for (int n = 0; n < 4; ++n)
#pragma unroll
      for (int v = 0; v < 8; ++v)
        out[(size_t)(rb + v) * ADIM + n * 16 + col] =
            (__bf16)(acc[n][v] * scale);
  } else {
    __bf16* out = Vt + (size_t)(h * BB + b) * ADIM * PP;  // (ad, p)
#pragma unroll
    for (int n = 0; n < 4; ++n)
#pragma unroll
      for (int v = 0; v < 8; ++v)
        out[(size_t)(n * 16 + col) * PP + p + rb + v] = (__bf16)acc[n][v];
  }
}

// ---------------------------------------------------------------------------
// Stage 2: attention with softmax over the BATCH axis (reference quirk).
// grid = (PP/16, H); block = 512 = 16 waves, wave w == batch b.
// Score tiles -> LDS, cross-batch max/sum-exp, normalized attw rebuilt as an
// A-fragment, WMMA'd against Vt.  Residual fused: x1 = att + x.
// ---------------------------------------------------------------------------
__global__ __launch_bounds__(512) void attention_kernel(
    const __bf16* __restrict__ Q, const __bf16* __restrict__ K,
    const __bf16* __restrict__ Vt, const float* __restrict__ x,
    float* __restrict__ x1) {
  __shared__ float Sl[16][16][32];  // [batch][p][q]  32 KB
  __shared__ float Ml[16][32];
  __shared__ float Dl[16][32];

  int p0 = blockIdx.x * 16;
  int h  = blockIdx.y;
  int wave = threadIdx.x >> 5;  // == batch index
  int lane = threadIdx.x & 31;
  int col = lane & 15, rb = (lane >> 4) * 8;
  int row = lane & 15, kb8 = (lane >> 4) * 8;

  const __bf16* Qhb = Q + ((size_t)(h * BB + wave) * PP + p0) * ADIM;
  const __bf16* Khb = K + (size_t)(h * BB + wave) * PP * ADIM;
  const __bf16* Vhb = Vt + (size_t)(h * BB + wave) * ADIM * PP;

  v16bf aq0 = load_a_frag(Qhb, ADIM);       // ad 0..31 (scale pre-folded)
  v16bf aq1 = load_a_frag(Qhb + 32, ADIM);  // ad 32..63
  v8f acc[4] = {};

  for (int q0 = 0; q0 < PP; q0 += 32) {
    // ---- scores: S = (scale*Q) K^T  (row-major K is the (N,K) B) ----
#pragma unroll
    for (int qs = 0; qs < 2; ++qs) {
      const __bf16* kbase = Khb + (size_t)(q0 + qs * 16) * ADIM;
      v8f s = {};
      s = wmma_bf16(aq0, load_b_frag(kbase, ADIM), s);
      s = wmma_bf16(aq1, load_b_frag(kbase + 32, ADIM), s);
#pragma unroll
      for (int v = 0; v < 8; ++v) Sl[wave][rb + v][qs * 16 + col] = s[v];
    }
    __syncthreads();

    // ---- softmax over batch: one thread per (p,q) slot (16*32 = 512) ----
    {
      int sp = threadIdx.x >> 5;
      int sq = threadIdx.x & 31;
      float m = Sl[0][sp][sq];
#pragma unroll
      for (int w = 1; w < 16; ++w) m = fmaxf(m, Sl[w][sp][sq]);
      float d = 0.f;
#pragma unroll
      for (int w = 0; w < 16; ++w) d += __expf(Sl[w][sp][sq] - m);
      Ml[sp][sq] = m;
      Dl[sp][sq] = d;
    }
    __syncthreads();

    // ---- rebuild attw as bf16 A-fragment (16x32 over q) ----
    v16bf aw;
#pragma unroll
    for (int j = 0; j < 16; ++j) {
      int k = kb8 + (j & 7) + (j >> 3) * 16;
      float w = __expf(Sl[wave][row][k] - Ml[row][k]) / Dl[row][k];
      aw[j] = (__bf16)w;
    }

    // ---- att += attw @ V   (Vt (ad,p) is the (N,K)-major B) ----
#pragma unroll
    for (int n = 0; n < 4; ++n) {
      v16bf bv = load_b_frag(Vhb + (size_t)(n * 16) * PP + q0, PP);
      acc[n] = wmma_bf16(aw, bv, acc[n]);
    }
    __syncthreads();
  }

  // ---- epilogue: x1[b][p][h*64+ad] = att + x ----
#pragma unroll
  for (int n = 0; n < 4; ++n)
#pragma unroll
    for (int v = 0; v < 8; ++v) {
      size_t idx = ((size_t)wave * PP + (p0 + rb + v)) * DM + h * ADIM +
                   n * 16 + col;
      x1[idx] = acc[n][v] + x[idx];
    }
}

// ---------------------------------------------------------------------------
// Stage 3: FFN GEMMs via the async-staged core.
// ---------------------------------------------------------------------------
// y1 = gelu(x1n @ W1 + b1), bf16 out.  M=16384 N=2048 K=512
__global__ __launch_bounds__(128) void ffn1_kernel(
    const __bf16* __restrict__ A, const __bf16* __restrict__ Bt /*(FF,D)*/,
    const float* __restrict__ bias, __bf16* __restrict__ Y) {
  __shared__ __bf16 Bs[2][64][BSTRIDE];
  int wave = threadIdx.x >> 5, lane = threadIdx.x & 31;
  size_t m0 = (size_t)blockIdx.x * 64 + wave * 16;
  size_t n0 = (size_t)blockIdx.y * 64;
  v8f acc[4] = {};
  staged_gemm_16x64(A, DM, Bt, DM, n0, m0, /*KC=*/16, Bs, acc);

  int col = lane & 15, rb = (lane >> 4) * 8;
#pragma unroll
  for (int n = 0; n < 4; ++n) {
    float bb = bias[n0 + n * 16 + col];
#pragma unroll
    for (int v = 0; v < 8; ++v) {
      float g = gelu_exact(acc[n][v] + bb);
      Y[(m0 + rb + v) * FFD + n0 + n * 16 + col] = (__bf16)g;
    }
  }
}

// out = gelu(y1 @ W2 + b2) + x1, f32 out.  M=16384 N=512 K=2048
__global__ __launch_bounds__(128) void ffn2_kernel(
    const __bf16* __restrict__ A, const __bf16* __restrict__ Bt /*(D,FF)*/,
    const float* __restrict__ bias, const float* __restrict__ x1,
    float* __restrict__ out) {
  __shared__ __bf16 Bs[2][64][BSTRIDE];
  int wave = threadIdx.x >> 5, lane = threadIdx.x & 31;
  size_t m0 = (size_t)blockIdx.x * 64 + wave * 16;
  size_t n0 = (size_t)blockIdx.y * 64;
  v8f acc[4] = {};
  staged_gemm_16x64(A, FFD, Bt, FFD, n0, m0, /*KC=*/64, Bs, acc);

  int col = lane & 15, rb = (lane >> 4) * 8;
#pragma unroll
  for (int n = 0; n < 4; ++n) {
    float bb = bias[n0 + n * 16 + col];
#pragma unroll
    for (int v = 0; v < 8; ++v) {
      size_t idx = (m0 + rb + v) * DM + n0 + n * 16 + col;
      out[idx] = gelu_exact(acc[n][v] + bb) + x1[idx];
    }
  }
}

// ---------------------------------------------------------------------------
// Host launcher
// ---------------------------------------------------------------------------
extern "C" void kernel_launch(void* const* d_in, const int* in_sizes, int n_in,
                              void* d_out, int out_size, void* d_ws,
                              size_t ws_size, hipStream_t stream) {
  (void)in_sizes; (void)n_in; (void)out_size; (void)ws_size;

  const float* x      = (const float*)d_in[0];
  const float* Wq     = (const float*)d_in[1];
  const float* Wk     = (const float*)d_in[2];
  const float* Wv     = (const float*)d_in[3];
  const float* gamma1 = (const float*)d_in[4];
  const float* beta1  = (const float*)d_in[5];
  const float* gamma2 = (const float*)d_in[6];
  const float* beta2  = (const float*)d_in[7];
  const float* W1     = (const float*)d_in[8];
  const float* b1     = (const float*)d_in[9];
  const float* W2     = (const float*)d_in[10];
  const float* b2     = (const float*)d_in[11];
  float* out = (float*)d_out;

  char* ws = (char*)d_ws;
  size_t off = 0;
  auto alloc = [&](size_t bytes) {
    size_t r = off;
    off = (off + bytes + 255) & ~(size_t)255;
    return r;
  };
  float*  psum  = (float*)(ws + alloc(1024 * 4));
  float*  psq   = (float*)(ws + alloc(1024 * 4));
  float*  stats = (float*)(ws + alloc(4 * 4));
  __bf16* xn    = (__bf16*)(ws + alloc((size_t)NX * 2));
  __bf16* wqt   = (__bf16*)(ws + alloc((size_t)HH * ADIM * DM * 2));
  __bf16* wkt   = (__bf16*)(ws + alloc((size_t)HH * ADIM * DM * 2));
  __bf16* wvt   = (__bf16*)(ws + alloc((size_t)HH * ADIM * DM * 2));
  __bf16* w1t   = (__bf16*)(ws + alloc((size_t)DM * FFD * 2));
  __bf16* w2t   = (__bf16*)(ws + alloc((size_t)FFD * DM * 2));
  __bf16* Qb    = (__bf16*)(ws + alloc((size_t)HH * MM * ADIM * 2));
  __bf16* Kb    = (__bf16*)(ws + alloc((size_t)HH * MM * ADIM * 2));
  __bf16* Vtb   = (__bf16*)(ws + alloc((size_t)HH * MM * ADIM * 2));
  float*  x1    = (float*)(ws + alloc((size_t)NX * 4));
  __bf16* x1n   = (__bf16*)(ws + alloc((size_t)NX * 2));
  __bf16* y1    = (__bf16*)(ws + alloc((size_t)MM * FFD * 2));

  // ---- global norm 1 ----
  reduce_partial_kernel<<<1024, 256, 0, stream>>>(x, NX, psum, psq);
  finalize_stats_kernel<<<1, 256, 0, stream>>>(psum, psq, 1024, (float)NX,
                                               stats);
  norm_to_bf16_kernel<<<4096, 256, 0, stream>>>(x, stats, gamma1, beta1, xn,
                                                NX);
  // ---- weight convert/transpose to (N,K)-major bf16 ----
  transpose_to_bf16_kernel<<<1024, 256, 0, stream>>>(Wq, wqt, HH, DM, ADIM);
  transpose_to_bf16_kernel<<<1024, 256, 0, stream>>>(Wk, wkt, HH, DM, ADIM);
  transpose_to_bf16_kernel<<<1024, 256, 0, stream>>>(Wv, wvt, HH, DM, ADIM);
  transpose_to_bf16_kernel<<<2048, 256, 0, stream>>>(W1, w1t, 1, DM, FFD);
  transpose_to_bf16_kernel<<<2048, 256, 0, stream>>>(W2, w2t, 1, FFD, DM);

  // ---- QKV projections (WMMA + async LDS staging) ----
  qkv_proj_kernel<<<dim3(MM / 64, HH, 3), 128, 0, stream>>>(
      xn, wqt, wkt, wvt, Qb, Kb, Vtb);

  // ---- attention (WMMA, softmax-over-batch, fused residual) ----
  attention_kernel<<<dim3(PP / 16, HH), 512, 0, stream>>>(Qb, Kb, Vtb, x, x1);

  // ---- global norm 2 ----
  reduce_partial_kernel<<<1024, 256, 0, stream>>>(x1, NX, psum, psq);
  finalize_stats_kernel<<<1, 256, 0, stream>>>(psum, psq, 1024, (float)NX,
                                               stats + 2);
  norm_to_bf16_kernel<<<4096, 256, 0, stream>>>(x1, stats + 2, gamma2, beta2,
                                                x1n, NX);

  // ---- FFN (WMMA + async LDS staging) ----
  ffn1_kernel<<<dim3(MM / 64, FFD / 64), 128, 0, stream>>>(x1n, w1t, b1, y1);
  ffn2_kernel<<<dim3(MM / 64, DM / 64), 128, 0, stream>>>(y1, w2t, b2, x1,
                                                          out);
}
